// Multi_Scale_Attention_old_17523466568046
// MI455X (gfx1250) — matrix operand order repaired and checked by reading, not compiled, not run on hardware
//
#include <hip/hip_runtime.h>
#include <hip/hip_bf16.h>

// Problem constants (match reference)
#define BB 256
#define CC 128
#define HH 8
#define WW 4
#define DD 4096
#define EPSLN 1e-6f

typedef __attribute__((ext_vector_type(16))) __bf16 v16bf;
typedef __attribute__((ext_vector_type(8)))  float  v8f;

union Frag16 { v16bf v; uint4 q[2]; };

__device__ __forceinline__ unsigned short f2bf(float f) {
  // round-to-nearest-even f32 -> bf16 (scalar path)
  unsigned int u = __float_as_uint(f);
  u += 0x7FFFu + ((u >> 16) & 1u);
  return (unsigned short)(u >> 16);
}

// Packed f32x2 -> bf16x2 in one VOP3: v_cvt_pk_bf16_f32 (RNE).
// The clang builtin is not declared on this toolchain, so emit the ISA op
// directly: dst[15:0] = bf16(src0), dst[31:16] = bf16(src1).
__device__ __forceinline__ unsigned pack2bf(float lo, float hi) {
  unsigned r;
  asm("v_cvt_pk_bf16_f32 %0, %1, %2" : "=v"(r) : "v"(lo), "v"(hi));
  return r;
}

__device__ __forceinline__ v8f v8zero() {
  v8f z = {0.f, 0.f, 0.f, 0.f, 0.f, 0.f, 0.f, 0.f};
  return z;
}

// CDNA5 async global->LDS copy (16B), tracked by ASYNCcnt (no VGPR round-trip,
// no LOADcnt/DScnt). Generic LDS pointer low 32 bits == LDS byte address
// (ISA 10.2 aperture rule), which is what VDST wants.
__device__ __forceinline__ void async_copy16(void* lds_dst, const void* gsrc) {
  asm volatile("global_load_async_to_lds_b128 %0, %1, off"
               :: "v"((unsigned)(unsigned long long)lds_dst), "v"(gsrc)
               : "memory");
}
__device__ __forceinline__ void wait_async0() {
#if defined(__has_builtin)
#if __has_builtin(__builtin_amdgcn_s_wait_asynccnt)
  __builtin_amdgcn_s_wait_asynccnt(0);
  return;
#endif
#endif
  asm volatile("s_wait_asynccnt 0" ::: "memory");
}

// ---------------------------------------------------------------------------
// Kernel 1: LayerNorm over D=4096, output packed bf16 (WMMA A operand).
// ---------------------------------------------------------------------------
__global__ __launch_bounds__(256) void ln_pack_kernel(
    const float* __restrict__ src, const float* __restrict__ g,
    const float* __restrict__ be, unsigned short* __restrict__ dst) {
  __shared__ float red[256];
  const int row = blockIdx.x, t = threadIdx.x;
  const float* p = src + (size_t)row * DD;
  float v[16];
  float s = 0.f;
#pragma unroll
  for (int i = 0; i < 16; ++i) { v[i] = p[t + i * 256]; s += v[i]; }
  red[t] = s; __syncthreads();
  for (int off = 128; off > 0; off >>= 1) { if (t < off) red[t] += red[t + off]; __syncthreads(); }
  const float mean = red[0] * (1.f / DD);
  __syncthreads();
  float sq = 0.f;
#pragma unroll
  for (int i = 0; i < 16; ++i) { float d = v[i] - mean; sq += d * d; }
  red[t] = sq; __syncthreads();
  for (int off = 128; off > 0; off >>= 1) { if (t < off) red[t] += red[t + off]; __syncthreads(); }
  const float rstd = rsqrtf(red[0] * (1.f / DD) + EPSLN);
  unsigned short* o = dst + (size_t)row * DD;
#pragma unroll
  for (int i = 0; i < 16; ++i) {
    const int col = t + i * 256;
    o[col] = f2bf((v[i] - mean) * rstd * g[col] + be[col]);
  }
}

// ---------------------------------------------------------------------------
// Kernel 4: final LayerNorm, f32 output.
// ---------------------------------------------------------------------------
__global__ __launch_bounds__(256) void ln_out_kernel(
    const float* __restrict__ src, const float* __restrict__ g,
    const float* __restrict__ be, float* __restrict__ dst) {
  __shared__ float red[256];
  const int row = blockIdx.x, t = threadIdx.x;
  const float* p = src + (size_t)row * DD;
  float v[16];
  float s = 0.f;
#pragma unroll
  for (int i = 0; i < 16; ++i) { v[i] = p[t + i * 256]; s += v[i]; }
  red[t] = s; __syncthreads();
  for (int off = 128; off > 0; off >>= 1) { if (t < off) red[t] += red[t + off]; __syncthreads(); }
  const float mean = red[0] * (1.f / DD);
  __syncthreads();
  float sq = 0.f;
#pragma unroll
  for (int i = 0; i < 16; ++i) { float d = v[i] - mean; sq += d * d; }
  red[t] = sq; __syncthreads();
  for (int off = 128; off > 0; off >>= 1) { if (t < off) red[t] += red[t + off]; __syncthreads(); }
  const float rstd = rsqrtf(red[0] * (1.f / DD) + EPSLN);
  float* o = dst + (size_t)row * DD;
#pragma unroll
  for (int i = 0; i < 16; ++i) {
    const int col = t + i * 256;
    o[col] = (v[i] - mean) * rstd * g[col] + be[col];
  }
}

// ---------------------------------------------------------------------------
// Kernel 2: four NT GEMMs  Out = A(256x4096,bf16) * W^T(4096x4096,f32) + bias
// grid = (N/128, 4). Full-M block tile (256x128): each HBM weight element is
// read exactly once -> HBM-bound floor ~11us for 256MB @ 23.3TB/s.
// Double-buffered LDS, one barrier per K-step:
//   issue async A->LDS(next) + W global loads(next) ; WMMA on LDS(cur) ;
//   convert+store W(next) ; s_wait_asynccnt 0 ; barrier.
// A staged via GLOBAL_LOAD_ASYNC_TO_LDS_B128 (ASYNCcnt); W converted with
// v_cvt_pk_bf16_f32. v_wmma_f32_16x16x32_bf16, 64x64 per wave.
// ---------------------------------------------------------------------------
#define PAG 40
#define PBG 40
__global__ __launch_bounds__(256) void gemm4_kernel(
    const unsigned short* __restrict__ nx, const unsigned short* __restrict__ nf,
    const unsigned short* __restrict__ nr,
    const float* __restrict__ W1, const float* __restrict__ W2,
    const float* __restrict__ Wf, const float* __restrict__ Wr,
    const float* __restrict__ b1, const float* __restrict__ b2,
    const float* __restrict__ bfv, const float* __restrict__ brv,
    float* __restrict__ o1, float* __restrict__ o2,
    float* __restrict__ of, float* __restrict__ orr) {
  __shared__ unsigned short smA[2][256 * PAG];  // 2 x 20KB
  __shared__ unsigned short smB[2][128 * PBG];  // 2 x 10KB

  const int gsel = blockIdx.y;
  const unsigned short* A = (gsel <= 1) ? nx : (gsel == 2 ? nf : nr);
  const float* Wm   = gsel == 0 ? W1 : gsel == 1 ? W2 : gsel == 2 ? Wf : Wr;
  const float* bias = gsel == 0 ? b1 : gsel == 1 ? b2 : gsel == 2 ? bfv : brv;
  float* Out        = gsel == 0 ? o1 : gsel == 1 ? o2 : gsel == 2 ? of : orr;

  const int nTile = blockIdx.x * 128;
  const int t = threadIdx.x, lane = t & 31, wid = t >> 5;
  const int l16 = lane & 15, half = lane >> 4;
  const int mBase = (wid >> 1) * 64;   // 4 waves along M
  const int nBase = (wid & 1) * 64;    // 2 waves along N

  // Per-thread staging coordinates (fixed across K-steps)
  const int rA = t >> 2, cA = (t & 3) * 8;          // A: 4 chunks/thread via +64 rows
  const int rW = t >> 3, cW = (t & 7) * 4;          // W: 4 chunks/thread via +32 rows

  v8f acc[4][4];
#pragma unroll
  for (int mi = 0; mi < 4; ++mi)
#pragma unroll
    for (int ni = 0; ni < 4; ++ni) acc[mi][ni] = v8zero();

  // ---- prologue: stage k0 = 0 into buffer 0 ----
#pragma unroll
  for (int i = 0; i < 4; ++i)
    async_copy16(&smA[0][(rA + i * 64) * PAG + cA],
                 A + (size_t)(rA + i * 64) * DD + cA);
  {
    float4 w[4];
#pragma unroll
    for (int i = 0; i < 4; ++i)
      w[i] = *(const float4*)(Wm + (size_t)(nTile + rW + i * 32) * DD + cW);
#pragma unroll
    for (int i = 0; i < 4; ++i) {
      uint2 hh;
      hh.x = pack2bf(w[i].x, w[i].y);
      hh.y = pack2bf(w[i].z, w[i].w);
      *(uint2*)&smB[0][(rW + i * 32) * PBG + cW] = hh;
    }
  }
  wait_async0();
  __syncthreads();

  for (int k0 = 0; k0 < DD; k0 += 32) {
    const int cur = (k0 >> 5) & 1, nxt = cur ^ 1;
    const bool hasNext = (k0 + 32) < DD;
    float4 w[4];

    if (hasNext) {
      const int kn = k0 + 32;
      // next A tile straight into LDS via the async engine
#pragma unroll
      for (int i = 0; i < 4; ++i)
        async_copy16(&smA[nxt][(rA + i * 64) * PAG + cA],
                     A + (size_t)(rA + i * 64) * DD + kn + cA);
      // next W tile into registers (needs f32->bf16 conversion)
#pragma unroll
      for (int i = 0; i < 4; ++i)
        w[i] = *(const float4*)(Wm + (size_t)(nTile + rW + i * 32) * DD + kn + cW);
      // stream-prefetch weights ~4 K-steps ahead into GL2
      if (k0 + 160 < DD)
        __builtin_prefetch(Wm + (size_t)(nTile + rW) * DD + k0 + 160 + cW, 0, 0);
    }

    // Fragment k order used by this code (wave32), same for A and B:
    // A 16x32: half 0 -> K 0..15 ; half 1 -> K 16..31 (contiguous 16)
    // B 32x16: half 0 -> K 0..15 ; half 1 -> K 16..31 (contiguous 16)
    Frag16 fa[4], fb[4];
#pragma unroll
    for (int mi = 0; mi < 4; ++mi) {
      const unsigned short* pA = &smA[cur][(mBase + mi * 16 + l16) * PAG + half * 16];
      fa[mi].q[0] = *(const uint4*)pA;
      fa[mi].q[1] = *(const uint4*)(pA + 8);
    }
#pragma unroll
    for (int ni = 0; ni < 4; ++ni) {
      const unsigned short* pB = &smB[cur][(nBase + ni * 16 + l16) * PBG + half * 16];
      fb[ni].q[0] = *(const uint4*)pB;
      fb[ni].q[1] = *(const uint4*)(pB + 8);
    }
#pragma unroll
    for (int mi = 0; mi < 4; ++mi)
#pragma unroll
      for (int ni = 0; ni < 4; ++ni)
        acc[mi][ni] = __builtin_amdgcn_wmma_f32_16x16x32_bf16(
            false, fa[mi].v, false, fb[ni].v, (short)0, acc[mi][ni], false, false);

    if (hasNext) {
#pragma unroll
      for (int i = 0; i < 4; ++i) {
        uint2 hh;
        hh.x = pack2bf(w[i].x, w[i].y);
        hh.y = pack2bf(w[i].z, w[i].w);
        *(uint2*)&smB[nxt][(rW + i * 32) * PBG + cW] = hh;
      }
    }
    wait_async0();     // next-A resident in LDS before the release barrier
    __syncthreads();
  }

  // Epilogue: C/D layout — lane<16: (M=v, N=lane); lane>=16: (M=v+8, N=lane-16)
#pragma unroll
  for (int ni = 0; ni < 4; ++ni) {
    const int n = nTile + nBase + ni * 16 + l16;
    const float bv = bias[n];
#pragma unroll
    for (int mi = 0; mi < 4; ++mi) {
#pragma unroll
      for (int v = 0; v < 8; ++v) {
        const int m = mBase + mi * 16 + v + half * 8;
        Out[(size_t)m * DD + n] = acc[mi][ni][v] + bv;
      }
    }
  }
}

// ---------------------------------------------------------------------------
// Kernel 3: per-(b,h) attention. One block per (b,h), 256 threads (8 waves).
// scores (K=4) scalar f32; softmax via wave32 shfl_xor; attn @ Ws^T via WMMA
// (bf16 in LDS, f32 accum); out = attn2 @ x1v scalar (K=128).
// Activation slices gathered with GLOBAL_LOAD_ASYNC_TO_LDS_B128.
// Dynamic LDS (~142KB) — within CDNA5's 320KB/WGP.
// ---------------------------------------------------------------------------
#define PTB 136  // bf16 pitch: 272B rows -> 16B aligned, banks r*68%64=4r distinct
#define POF 132  // f32 pitch
__global__ __launch_bounds__(256) void attn_kernel(
    const float* __restrict__ x1, const float* __restrict__ x2,
    const float* __restrict__ xfp, const float* __restrict__ xrp,
    const float* __restrict__ Ws, const float* __restrict__ bs,
    float* __restrict__ y) {
  extern __shared__ char smem[];
  float* sx1 = (float*)smem;       // 128 x 4
  float* sx2 = sx1 + 512;
  float* sxf = sx2 + 512;
  float* sxr = sxf + 512;
  unsigned short* sAt = (unsigned short*)(sxr + 512);  // attn bf16 128 x PTB
  unsigned short* sWs = sAt + 128 * PTB;               // Ws   bf16 128 x PTB
  float* sO = (float*)(sWs + 128 * PTB);               // attn2 f32 128 x POF

  const int bh = blockIdx.x, b = bh >> 3, h = bh & 7;
  const int t = threadIdx.x, lane = t & 31, wid = t >> 5;
  const int l16 = lane & 15, half = lane >> 4;

  const float* srcs[4] = { x1 + (size_t)b * DD, x2 + (size_t)b * DD,
                           xfp + (size_t)b * DD, xrp + (size_t)b * DD };
  float* dsts[4] = { sx1, sx2, sxf, sxr };
  // Gather the four (128 x 4) slices: element (c,w) at offset c*32 + h*4 + w.
  // 16B per row -> async global->LDS copies (ASYNCcnt).
#pragma unroll
  for (int i = 0; i < 2; ++i) {
    const int idx = t + i * 256;
    const int bsel = idx >> 7, c = idx & 127;
    async_copy16(dsts[bsel] + c * 4, srcs[bsel] + c * 32 + h * 4);
  }
  // Ws -> bf16 in LDS (128 x 128); needs conversion, so VGPR path
#pragma unroll
  for (int i = 0; i < 16; ++i) {
    const int idx = t + i * 256;
    const int r = idx >> 5, c4 = idx & 31;
    const float4 w = *(const float4*)(Ws + (size_t)r * CC + c4 * 4);
    uint2 hh;
    hh.x = pack2bf(w.x, w.y);
    hh.y = pack2bf(w.z, w.w);
    *(uint2*)&sWs[r * PTB + c4 * 4] = hh;
  }
  wait_async0();
  __syncthreads();

  // scores + softmax: wave wid owns rows [wid*16, wid*16+16)
  const float invs = 0.088388347648318447f;  // 1/sqrt(128)
  for (int rr = 0; rr < 16; ++rr) {
    const int c = wid * 16 + rr;
    const float a0 = sx1[c * 4 + 0], a1 = sx1[c * 4 + 1], a2 = sx1[c * 4 + 2], a3 = sx1[c * 4 + 3];
    const float e0 = sx2[c * 4 + 0], e1 = sx2[c * 4 + 1], e2 = sx2[c * 4 + 2], e3 = sx2[c * 4 + 3];
    float sc[4];
#pragma unroll
    for (int j = 0; j < 4; ++j) {
      const int k = lane + j * 32;
      const float* f = sxf + k * 4;
      const float* r = sxr + k * 4;
      sc[j] = (a0 * f[0] + a1 * f[1] + a2 * f[2] + a3 * f[3] +
               e0 * r[0] + e1 * r[1] + e2 * r[2] + e3 * r[3]) * invs;
    }
    float mx = fmaxf(fmaxf(sc[0], sc[1]), fmaxf(sc[2], sc[3]));
#pragma unroll
    for (int off = 16; off > 0; off >>= 1) mx = fmaxf(mx, __shfl_xor(mx, off, 32));
    float ssum = 0.f;
#pragma unroll
    for (int j = 0; j < 4; ++j) { sc[j] = __expf(sc[j] - mx); ssum += sc[j]; }
#pragma unroll
    for (int off = 16; off > 0; off >>= 1) ssum += __shfl_xor(ssum, off, 32);
    const float inv = 1.f / ssum;
#pragma unroll
    for (int j = 0; j < 4; ++j) sAt[c * PTB + lane + j * 32] = f2bf(sc[j] * inv);
  }
  // No barrier needed: each wave's WMMA A-frags read only the 16 rows that
  // same wave just wrote (DS ops are in-order per wave); sWs synced earlier.

  const int mBase = wid * 16;
  v8f acc[8];
#pragma unroll
  for (int ni = 0; ni < 8; ++ni) acc[ni] = v8zero();
  for (int k0 = 0; k0 < 128; k0 += 32) {
    Frag16 fa;
    const unsigned short* pA = &sAt[(mBase + l16) * PTB + k0 + half * 16];
    fa.q[0] = *(const uint4*)pA;
    fa.q[1] = *(const uint4*)(pA + 8);
#pragma unroll
    for (int ni = 0; ni < 8; ++ni) {
      Frag16 fb;
      const unsigned short* pB = &sWs[(ni * 16 + l16) * PTB + k0 + half * 16];
      fb.q[0] = *(const uint4*)pB;
      fb.q[1] = *(const uint4*)(pB + 8);
      acc[ni] = __builtin_amdgcn_wmma_f32_16x16x32_bf16(
          false, fa.v, false, fb.v, (short)0, acc[ni], false, false);
    }
  }
#pragma unroll
  for (int ni = 0; ni < 8; ++ni) {
    const int n = ni * 16 + l16;
    const float bv = bs[n];
#pragma unroll
    for (int v = 0; v < 8; ++v)
      sO[(mBase + v + half * 8) * POF + n] = acc[ni][v] + bv;
  }
  __syncthreads();

  // out[c][w] = sum_k attn2[c][k] * x1v[k][w]; store in (c*32 + h*4 + w) order
  float* yb = y + (size_t)b * DD;
#pragma unroll
  for (int i = 0; i < 2; ++i) {
    const int idx = t + i * 256;  // 0..511
    const int c = idx >> 2, w = idx & 3;
    float s = 0.f;
    for (int k = 0; k < 128; ++k) s += sO[c * POF + k] * sx1[k * 4 + w];
    yb[c * 32 + h * 4 + w] = s;
  }
}

// ---------------------------------------------------------------------------
extern "C" void kernel_launch(void* const* d_in, const int* in_sizes, int n_in,
                              void* d_out, int out_size, void* d_ws, size_t ws_size,
                              hipStream_t stream) {
  (void)in_sizes; (void)n_in; (void)out_size; (void)ws_size;
  const float* x    = (const float*)d_in[0];
  const float* xf   = (const float*)d_in[1];
  const float* xr   = (const float*)d_in[2];
  const float* g_x  = (const float*)d_in[3];
  const float* be_x = (const float*)d_in[4];
  const float* g_f  = (const float*)d_in[5];
  const float* be_f = (const float*)d_in[6];
  const float* g_r  = (const float*)d_in[7];
  const float* be_r = (const float*)d_in[8];
  const float* g_2  = (const float*)d_in[9];
  const float* be_2 = (const float*)d_in[10];
  const float* W1   = (const float*)d_in[11];
  const float* b1   = (const float*)d_in[12];
  const float* W2   = (const float*)d_in[13];
  const float* b2   = (const float*)d_in[14];
  const float* Wf   = (const float*)d_in[15];
  const float* bfv  = (const float*)d_in[16];
  const float* Wr   = (const float*)d_in[17];
  const float* brv  = (const float*)d_in[18];
  const float* Ws   = (const float*)d_in[19];
  const float* bs   = (const float*)d_in[20];

  const size_t row = (size_t)BB * DD;
  unsigned short* nxB = (unsigned short*)d_ws;  // 2MB each
  unsigned short* nfB = nxB + row;
  unsigned short* nrB = nfB + row;
  float* x1  = (float*)(nrB + row);             // 4MB each
  float* x2  = x1 + row;
  float* xfp = x2 + row;
  float* xrp = xfp + row;
  float* y   = xrp + row;                       // pre-LN output, 4MB

  ln_pack_kernel<<<BB, 256, 0, stream>>>(x,  g_x, be_x, nxB);
  ln_pack_kernel<<<BB, 256, 0, stream>>>(xf, g_f, be_f, nfB);
  ln_pack_kernel<<<BB, 256, 0, stream>>>(xr, g_r, be_r, nrB);

  gemm4_kernel<<<dim3(32, 4), 256, 0, stream>>>(
      nxB, nfB, nrB, W1, W2, Wf, Wr, b1, b2, bfv, brv, x1, x2, xfp, xrp);

  const size_t smemBytes = (size_t)4 * 512 * 4           // x1/x2/xf/xr slices
                         + (size_t)2 * 128 * PTB * 2     // attn + Ws (bf16)
                         + (size_t)128 * POF * 4;        // attn2 (f32)
  attn_kernel<<<BB * HH, 256, smemBytes, stream>>>(x1, x2, xfp, xrp, Ws, bs, y);

  ln_out_kernel<<<BB, 256, 0, stream>>>(y, g_2, be_2, (float*)d_out);
}
